// GaussianSelfAttention_11003706212621
// MI455X (gfx1250) — compile-verified, matrix operand order checked by reading, not run on hardware
//
#include <hip/hip_runtime.h>
#include <cstddef>
#include <cstdint>

// ---------------------------------------------------------------------------
// Gaussian self-attention for MI455X (gfx1250, wave32, WMMA + TDM).
//   B=32, H=W=32 (q/k grid of 1024), D=128, HEADS=16.
//   ~155 GFLOP vs ~100 MB mandatory HBM traffic -> run GEMMs on
//   v_wmma_f32_16x16x32_f16 (f32 accum) and stage tiles with the Tensor Data
//   Mover (tensor_load_to_lds, TENSORcnt), double-buffered.
// Workspace layout (~40.5 MB, all L2-resident on the 192MB L2):
//   [0,       32MB)  Pws : f16 probs, [head][q][k]        (16 x 1024 x 1024)
//   [32MB,    40MB)  Xt  : f16 hidden^T, [b*128+d][kl]    (4096 x 1024)
//   [40MB, 40.5MB)   Whh : f16 W^T per head, [h][dout][d] (16 x 128 x 128)
//   [40.5MB, ...)    uws : f32 u vectors, [16][5]
// ---------------------------------------------------------------------------

typedef _Float16 v16h __attribute__((ext_vector_type(16)));
typedef _Float16 v8h  __attribute__((ext_vector_type(8)));
typedef float    v8f  __attribute__((ext_vector_type(8)));
typedef float    v4f  __attribute__((ext_vector_type(4)));

#define NQ      1024      // H*W query/key positions
#define NHEADS  16
#define DMODEL  128
#define NB      32

// ---- tiny prep kernels ----------------------------------------------------

__global__ void u_kernel(const float* __restrict__ centers,
                         const float* __restrict__ spreads,
                         float* __restrict__ uws) {
  int h = threadIdx.x;
  if (h >= NHEADS) return;
  const float* S = spreads + h * 4;              // row-major 2x2
  float a  = S[0]*S[0] + S[1]*S[1];              // inv_cov[0][0]
  float bb = S[0]*S[2] + S[1]*S[3];              // inv_cov[0][1]
  float c  = S[2]*S[2] + S[3]*S[3];              // inv_cov[1][1]
  float mu1 = centers[h*2+0], mu2 = centers[h*2+1];
  uws[h*5+0] = a*mu1 + bb*mu2;
  uws[h*5+1] = c*mu2 + bb*mu1;
  uws[h*5+2] = -0.5f*a;
  uws[h*5+3] = -0.5f*c;
  uws[h*5+4] = -bb;
}

// hidden (b, kl, d) f32 -> Xt[(b*128+d)][kl] f16  (8MB, stays in L2)
__global__ void xt_kernel(const float* __restrict__ hidden,
                          _Float16* __restrict__ Xt) {
  int idx = blockIdx.x * 256 + threadIdx.x;      // < 32*1024*128
  int b  = idx >> 17;
  int r  = idx & 131071;
  int kl = r >> 7;
  int d  = r & 127;
  Xt[(size_t)(b*128 + d) * NQ + kl] = (_Float16)hidden[idx];
}

// value_weight (dout, h*128+d) f32 -> Whh[h][dout][d] f16
__global__ void wh_kernel(const float* __restrict__ vw,
                          _Float16* __restrict__ Whh) {
  int idx = blockIdx.x * 256 + threadIdx.x;      // < 128*2048
  int dout = idx >> 11;
  int c    = idx & 2047;
  int h    = c >> 7;
  int d    = c & 127;
  Whh[((size_t)h * 128 + dout) * 128 + d] = (_Float16)vw[idx];
}

// ---- softmax over 1024 keys per (q, head); one wave each ------------------
// Lane owns one full key row k == lane (32 contiguous l values) so the 64MB
// f32 + 32MB f16 probs writes vectorize to b128 stores (this kernel IS the
// HBM roofline; everything else is L2 traffic).

__global__ __launch_bounds__(256) void probs_kernel(
    const float* __restrict__ uws,
    float* __restrict__ probs_out,        // d_out + 4194304, [(q*16+h)*1024+kl]
    _Float16* __restrict__ Pws) {         // [h][q][k]
  int gtid = blockIdx.x * 256 + threadIdx.x;
  int wid  = gtid >> 5;                   // 0..16383 == q*16 + h
  int lane = threadIdx.x & 31;
  int h = wid & 15;
  int q = wid >> 4;
  int i = q >> 5, j = q & 31;

  float u0 = uws[h*5+0], u1 = uws[h*5+1], u2 = uws[h*5+2];
  float u3 = uws[h*5+3], u4 = uws[h*5+4];

  // key row k == lane: dk constant per lane; s = c0 + c1*dl + u3*dl^2
  float dk = (float)(lane - i);
  float c0 = u0*dk + u2*dk*dk;
  float c1 = u1 + u4*dk;

  float sc[32];
  float m = -1e30f;
#pragma unroll
  for (int t = 0; t < 32; ++t) {
    float dl = (float)(t - j);
    float s = c0 + c1*dl + u3*dl*dl;
    sc[t] = s;
    m = fmaxf(m, s);
  }
#pragma unroll
  for (int off = 16; off > 0; off >>= 1) m = fmaxf(m, __shfl_xor(m, off, 32));

  float sum = 0.f;
#pragma unroll
  for (int t = 0; t < 32; ++t) { float e = __expf(sc[t] - m); sc[t] = e; sum += e; }
#pragma unroll
  for (int off = 16; off > 0; off >>= 1) sum += __shfl_xor(sum, off, 32);
  float inv = 1.f / sum;

  float*    po = probs_out + (size_t)wid * NQ + lane * 32;   // 128B aligned
  _Float16* ph = Pws + ((size_t)h << 20) + (size_t)q * NQ + lane * 32;
#pragma unroll
  for (int t = 0; t < 32; t += 4) {
    v4f p4 = { sc[t]*inv, sc[t+1]*inv, sc[t+2]*inv, sc[t+3]*inv };
    *(v4f*)(po + t) = p4;
  }
#pragma unroll
  for (int t = 0; t < 32; t += 8) {
    v8h p8;
#pragma unroll
    for (int e = 0; e < 8; ++e) p8[e] = (_Float16)(sc[t+e] * inv);
    *(v8h*)(ph + t) = p8;
  }
}

// ---- fragment load matching the ISA 16-bit A/B operand layout -------------
// lane<16:  row = lane,     K = 0..7 and 16..23
// lane>=16: row = lane-16,  K = 8..15 and 24..31
__device__ __forceinline__ v16h load_frag(const _Float16* row, int half) {
  union { v16h v; v8h h2[2]; } u;
  u.h2[0] = *(const v8h*)(row + 8*half);
  u.h2[1] = *(const v8h*)(row + 16 + 8*half);
  return u.v;
}

// ---- Tensor Data Mover: 2D f16 tile (rows x 32) -> LDS with stride-40 pad -
#define LDA 40    // f16 stride for 32-wide K tiles (pad kills bank conflicts)
#define LDC 136   // f16 stride for the 128-wide C1 tile

#if defined(__gfx1250__) && __has_builtin(__builtin_amdgcn_tensor_load_to_lds)
#define USE_TDM 1
typedef unsigned int u32x4 __attribute__((ext_vector_type(4)));
typedef int          i32x4 __attribute__((ext_vector_type(4)));
typedef int          i32x8 __attribute__((ext_vector_type(8)));

// D# groups per CDNA5 ISA:
//  g0: count=1 | lds_addr | global_addr[56:0] | type=2
//  g1: data_size=2B, pad_enable, pad_interval=16 DW (one 32-f16 row),
//      pad_amount=4 DW (16B) -> LDS row stride 80B == LDA f16,
//      tensor_dim0=tile_dim0=32, tensor_dim1=tile_dim1=rows,
//      tensor_dim0_stride=1024 elements.
__device__ __forceinline__ void tdm_load_2d(unsigned lds_off,
                                            const _Float16* gptr, int rows) {
  uint64_t ga = (uint64_t)(uintptr_t)gptr;
  u32x4 g0;
  g0[0] = 1u;                                            // count=1
  g0[1] = lds_off;                                       // lds_addr
  g0[2] = (unsigned)(ga & 0xFFFFFFFFu);                  // global_addr lo
  g0[3] = (unsigned)((ga >> 32) & 0x01FFFFFFu)           // global_addr hi
        | 0x80000000u;                                   // type=2 ("image")
  i32x8 g1;
  g1[0] = (1 << 16) | (1 << 20) | (3 << 22) | (3 << 25); // dsz/pad fields
  g1[1] = (32 << 16);                                    // tensor_dim0 lo16
  g1[2] = (rows << 16);                                  // tensor_dim1 lo16
  g1[3] = (32 << 16);                                    // tile_dim0
  g1[4] = rows;                                          // tile_dim1
  g1[5] = 1024;                                          // dim0_stride lo32
  g1[6] = 0;
  g1[7] = 0;
  i32x4 z4 = {0, 0, 0, 0};
#if __clang_major__ >= 23
  i32x8 z8 = {0, 0, 0, 0, 0, 0, 0, 0};
  __builtin_amdgcn_tensor_load_to_lds(g0, g1, z4, z4, z8, 0);
#else
  __builtin_amdgcn_tensor_load_to_lds(g0, g1, z4, z4, 0);
#endif
}
#else
#define USE_TDM 0
#endif

// ---- fused GEMM1 (P_h @ X_b) + GEMM2 (C1 @ W_h^T), head loop in-register --
// grid: 512 = qt(16) x bb(32).  Tile: 64 q-rows x 128 cols.  8 waves, each
// a 32x32 subtile.  A/B K-slices double-buffered via TDM (wave 0 issues,
// TENSORcnt pipelines: wait<=2 means "previous slice landed").

__global__ __launch_bounds__(256, 1) void gemm_kernel(
    const _Float16* __restrict__ Pws,
    const _Float16* __restrict__ Xt,
    const _Float16* __restrict__ Whh,
    float* __restrict__ out) {
  __shared__ __align__(16) _Float16 sA[2][64  * LDA];  // P tile  (64 q x 32 k)
  __shared__ __align__(16) _Float16 sB[2][128 * LDA];  // X tile^T (128 n x 32 k)
  __shared__ __align__(16) _Float16 sC[64 * LDC];      // C1 f16  (64 q x 128 d)

  int tid  = threadIdx.x;
  int wave = tid >> 5, lane = tid & 31;
  int wm = wave & 1, wn = wave >> 1;
  int r = lane & 15, half = lane >> 4;

  int qt = blockIdx.x >> 5;        // 0..15
  int bb = blockIdx.x & 31;        // 0..31
  int qbase = qt * 64;

  const _Float16* Xb = Xt + (size_t)bb * 128 * NQ;

#if USE_TDM
  unsigned ldsA0 = (unsigned)(uintptr_t)&sA[0][0];
  unsigned ldsA1 = (unsigned)(uintptr_t)&sA[1][0];
  unsigned ldsB0 = (unsigned)(uintptr_t)&sB[0][0];
  unsigned ldsB1 = (unsigned)(uintptr_t)&sB[1][0];
#endif

  v8f acc2[2][2];
#pragma unroll
  for (int mi = 0; mi < 2; ++mi)
#pragma unroll
    for (int ni = 0; ni < 2; ++ni)
#pragma unroll
      for (int e = 0; e < 8; ++e) acc2[mi][ni][e] = 0.f;

  for (int h = 0; h < NHEADS; ++h) {
    const _Float16* Ph = Pws + ((size_t)h << 20) + (size_t)qbase * NQ;

    v8f acc[2][2];
#pragma unroll
    for (int mi = 0; mi < 2; ++mi)
#pragma unroll
      for (int ni = 0; ni < 2; ++ni)
#pragma unroll
        for (int e = 0; e < 8; ++e) acc[mi][ni][e] = 0.f;

#if USE_TDM
    if (wave == 0) {                       // prologue: slice 0 into buffer 0
      tdm_load_2d(ldsA0, Ph, 64);
      tdm_load_2d(ldsB0, Xb, 128);
    }
#endif
    for (int t = 0; t < 32; ++t) {
      int kk = t * 32;
      int buf;
#if USE_TDM
      buf = t & 1;
      if (wave == 0) {
        if (t + 1 < 32) {                  // issue next slice, other buffer
          tdm_load_2d((t & 1) ? ldsA0 : ldsA1, Ph + kk + 32, 64);
          tdm_load_2d((t & 1) ? ldsB0 : ldsB1, Xb + kk + 32, 128);
          __builtin_amdgcn_s_wait_tensorcnt(2);  // slice t landed (in-order)
        } else {
          __builtin_amdgcn_s_wait_tensorcnt(0);
        }
      }
      __syncthreads();
      asm volatile("" ::: "memory");       // LDS written by TDM, not by stores
#else
      buf = 0;
      __syncthreads();
      {
        int rowA = tid >> 2, kcA = (tid & 3) << 3;
        *(v8h*)(&sA[0][rowA * LDA + kcA]) =
            *(const v8h*)(Ph + (size_t)rowA * NQ + kk + kcA);
#pragma unroll
        for (int cc = 0; cc < 2; ++cc) {
          int c = tid * 2 + cc;
          int rowB = c >> 2, kcB = (c & 3) << 3;
          *(v8h*)(&sB[0][rowB * LDA + kcB]) =
              *(const v8h*)(Xb + (size_t)rowB * NQ + kk + kcB);
        }
        if (kk + 32 < NQ) {
          __builtin_prefetch(Ph + (size_t)(tid >> 2) * NQ + kk + 32, 0, 0);
          __builtin_prefetch(Xb + (size_t)(tid >> 1) * NQ + kk + 32, 0, 0);
        }
      }
      __syncthreads();
#endif
      v16h afr[2], bfr[2];
#pragma unroll
      for (int mi = 0; mi < 2; ++mi)
        afr[mi] = load_frag(&sA[buf][(wm*32 + mi*16 + r) * LDA], half);
#pragma unroll
      for (int ni = 0; ni < 2; ++ni)
        bfr[ni] = load_frag(&sB[buf][(wn*32 + ni*16 + r) * LDA], half);
#pragma unroll
      for (int mi = 0; mi < 2; ++mi)
#pragma unroll
        for (int ni = 0; ni < 2; ++ni)
          acc[mi][ni] = __builtin_amdgcn_wmma_f32_16x16x32_f16(
              false, afr[mi], false, bfr[ni], (short)0, acc[mi][ni],
              false, false);
      __syncthreads();                     // all reads of buf done before reuse
    }

    // C/D layout: vgpr rr -> M = rr + 8*half (per 16x16 subtile), N = lane%16
#pragma unroll
    for (int mi = 0; mi < 2; ++mi)
#pragma unroll
      for (int ni = 0; ni < 2; ++ni)
#pragma unroll
        for (int rr = 0; rr < 8; ++rr) {
          int M = wm*32 + mi*16 + rr + 8*half;
          int N = wn*32 + ni*16 + r;
          sC[M * LDC + N] = (_Float16)acc[mi][ni][rr];
        }
    __syncthreads();

    // GEMM2: acc2 += C1(64x128) @ W_h^T, W stored [dout][d] row-major (L2 hit)
    const _Float16* Wh = Whh + (size_t)h * 128 * 128;
#pragma unroll
    for (int ks = 0; ks < 128; ks += 32) {
      v16h a2[2], b2[2];
#pragma unroll
      for (int mi = 0; mi < 2; ++mi)
        a2[mi] = load_frag(&sC[(wm*32 + mi*16 + r) * LDC + ks], half);
#pragma unroll
      for (int ni = 0; ni < 2; ++ni)
        b2[ni] = load_frag(Wh + (size_t)(wn*32 + ni*16 + r) * 128 + ks, half);
#pragma unroll
      for (int mi = 0; mi < 2; ++mi)
#pragma unroll
        for (int ni = 0; ni < 2; ++ni)
          acc2[mi][ni] = __builtin_amdgcn_wmma_f32_16x16x32_f16(
              false, a2[mi], false, b2[ni], (short)0, acc2[mi][ni],
              false, false);
    }
    __syncthreads();                       // sC free for next head
  }

  // single deterministic store: out[b][q][dout]
  float* outb = out + ((size_t)bb * NQ + qbase) * DMODEL;
#pragma unroll
  for (int mi = 0; mi < 2; ++mi)
#pragma unroll
    for (int ni = 0; ni < 2; ++ni)
#pragma unroll
      for (int rr = 0; rr < 8; ++rr) {
        int M = wm*32 + mi*16 + rr + 8*half;
        int N = wn*32 + ni*16 + r;
        outb[(size_t)M * DMODEL + N] = acc2[mi][ni][rr];
      }
}

// ---------------------------------------------------------------------------

extern "C" void kernel_launch(void* const* d_in, const int* in_sizes, int n_in,
                              void* d_out, int out_size, void* d_ws, size_t ws_size,
                              hipStream_t stream) {
  const float* hidden  = (const float*)d_in[0];   // (32,32,32,128)
  const float* centers = (const float*)d_in[1];   // (16,2)
  const float* spreads = (const float*)d_in[2];   // (16,2,2)
  const float* vw      = (const float*)d_in[3];   // (128,2048)
  float* out = (float*)d_out;                     // out[4194304] ++ probs[16777216]

  char* ws = (char*)d_ws;
  _Float16* Pws = (_Float16*)ws;                              // 32 MB
  _Float16* Xt  = (_Float16*)(ws + ((size_t)32 << 20));       //  8 MB
  _Float16* Whh = (_Float16*)(ws + ((size_t)40 << 20));       // 512 KB
  float*    uws = (float*)(ws + ((size_t)40 << 20) + (1u << 19));

  u_kernel    <<<1,     32,  0, stream>>>(centers, spreads, uws);
  xt_kernel   <<<16384, 256, 0, stream>>>(hidden, Xt);
  wh_kernel   <<<1024,  256, 0, stream>>>(vw, Whh);
  probs_kernel<<<2048,  256, 0, stream>>>(uws, out + 4194304, Pws);
  gemm_kernel <<<512,   256, 0, stream>>>(Pws, Xt, Whh, out);
}
